// RippleNetPlus_71863392797050
// MI455X (gfx1250) — matrix operand order repaired and checked by reading, not compiled, not run on hardware
//
#include <hip/hip_runtime.h>
#include <hip/hip_bf16.h>

typedef __attribute__((ext_vector_type(16))) _Float16 v16h;
typedef __attribute__((ext_vector_type(8)))  float    v8f;

#define DIM   32
#define NHOP  2
#define NMEM  32
#define NB    4      // batch rows per workgroup
#define NWAVE 8

__device__ __forceinline__ float sigmoidf_(float x) { return 1.0f / (1.0f + __expf(-x)); }

// CDNA5 has a native TANH transcendental (TRANS32 class). Use the builtin when
// the toolchain exposes it; otherwise a branch-free exp-based form that lowers
// to a single v_exp_f32 plus a few VALU ops (no EXEC-masked range splits).
__device__ __forceinline__ float fast_tanhf_(float x) {
#if __has_builtin(__builtin_amdgcn_tanhf)
    return __builtin_amdgcn_tanhf(x);
#else
    return 1.0f - 2.0f / (1.0f + __expf(2.0f * x));
#endif
}

__global__ __launch_bounds__(256)
void ripplenet_kernel(const int* __restrict__ h_i,
                      const int* __restrict__ R_i,
                      const int* __restrict__ t_i,
                      const int* __restrict__ v_i,
                      const float* __restrict__ entity_emb,
                      const float* __restrict__ relation_emb,
                      const float* __restrict__ agg_w1,
                      const float* __restrict__ agg_b1,
                      const float* __restrict__ agg_w2,
                      const float* __restrict__ agg_b2,
                      const float* __restrict__ gru_w_ih,
                      const float* __restrict__ gru_w_hh,
                      const float* __restrict__ gru_b_ih,
                      const float* __restrict__ gru_b_hh,
                      const float* __restrict__ ans_w,
                      const float* __restrict__ ans_b,
                      float* __restrict__ out)
{
    __shared__ _Float16 zT[NWAVE][16][128];     // 32 KB: per-wave 16x128 f16 A-tiles
    __shared__ float    tS[NB][NMEM][DIM];      // 16 KB: tail-entity cache
    __shared__ _Float16 w1h[DIM * 4 * DIM];     //  8 KB: agg_w1 in f16, [n][k] row-major
    __shared__ float    vS[NB][DIM];
    __shared__ float    MS[NB][DIM];
    __shared__ float    lg[NB][NMEM];

    const int tid  = threadIdx.x;
    const int wave = tid >> 5;
    const int lane = tid & 31;

    // ---- init: stage agg_w1 as f16; load v embeddings; M = v ----
    for (int i = tid; i < DIM * 4 * DIM; i += 256) w1h[i] = (_Float16)agg_w1[i];
    if (wave < NB) {
        const int bg = blockIdx.x * NB + wave;
        const float vv = entity_emb[(size_t)v_i[bg] * DIM + lane];
        vS[wave][lane] = vv;
        MS[wave][lane] = vv;
    }
    __syncthreads();

    // ---- pre-build B fragments for v_wmma_f32_16x16x32_f16 ----
    // B is K(32) x N(16): lane&15 = column N; element h -> K=(h<8?h:h+8) + 8*(lane>=16).
    const int bcol = lane & 15;
    const int bko  = (lane >> 4) << 4;
    v16h B[2][4];
#pragma unroll
    for (int nt = 0; nt < 2; ++nt)
#pragma unroll
        for (int kt = 0; kt < 4; ++kt)
#pragma unroll
            for (int hh = 0; hh < 16; ++hh)
                B[nt][kt][hh] = w1h[(nt * 16 + bcol) * 128 + kt * 32 + (hh < 8 ? hh : hh + 8) + (bko >> 1)];

    const float b1A = agg_b1[bcol];
    const float b1B = agg_b1[bcol + 16];
    const float w2A = agg_w2[bcol];
    const float w2B = agg_w2[bcol + 16];
    const float b2  = agg_b2[0];

    for (int hop = 0; hop < NHOP; ++hop) {
        // ---- phase 1: per-wave, 16 (b,m) pairs: Rh matvec + f16 feature tile ----
        for (int r = 0; r < 16; ++r) {
            const int pair = wave * 16 + r;          // 0..127
            const int bl   = pair >> 5;              // local batch row
            const int m    = pair & 31;              // memory slot
            const int bg   = blockIdx.x * NB + bl;
            const int ib   = (bg * NHOP + hop) * NMEM + m;
            const int hidx = h_i[ib];
            const int tidx = t_i[ib];
            const int ridx = R_i[ib];

            const float hv = entity_emb[(size_t)hidx * DIM + lane];
            const float4* Rrow =
                (const float4*)(relation_emb + (size_t)ridx * DIM * DIM + lane * DIM);
            float acc = 0.0f;
#pragma unroll
            for (int c = 0; c < 8; ++c) {
                float4 q = Rrow[c];
                acc += q.x * __shfl(hv, 4 * c + 0, 32);
                acc += q.y * __shfl(hv, 4 * c + 1, 32);
                acc += q.z * __shfl(hv, 4 * c + 2, 32);
                acc += q.w * __shfl(hv, 4 * c + 3, 32);
            }
            tS[bl][m][lane] = entity_emb[(size_t)tidx * DIM + lane];

            const float vv = vS[bl][lane];
            const float Mv = MS[bl][lane];
            zT[wave][r][lane]      = (_Float16)(acc * vv);
            zT[wave][r][32 + lane] = (_Float16)(acc * Mv);
            zT[wave][r][64 + lane] = (_Float16)fabsf(acc - vv);
            zT[wave][r][96 + lane] = (_Float16)fabsf(acc - Mv);
        }

        // ---- phase 2: WMMA  z(16x128) @ agg_w1^T(128x32), f32 accumulate ----
        // A layout: row = lane&15; half h -> K=(h<8?h:h+8) + 8*(lane>=16), per 32-K chunk.
        const int arow  = lane & 15;
        const int akoff = (lane >> 4) << 3;
        v8f acc0 = {};
        v8f acc1 = {};
#pragma unroll
        for (int kt = 0; kt < 4; ++kt) {
            v16h a;
#pragma unroll
            for (int hh = 0; hh < 16; ++hh) {
                const int lk = (hh < 8 ? hh : hh + 8) + akoff;
                a[hh] = zT[wave][arow][kt * 32 + lk];
            }
            acc0 = __builtin_amdgcn_wmma_f32_16x16x32_f16(false, a, false, B[0][kt],
                                                          (short)0, acc0, false, false);
            acc1 = __builtin_amdgcn_wmma_f32_16x16x32_f16(false, a, false, B[1][kt],
                                                          (short)0, acc1, false, false);
        }

        // ---- phase 3: tanh + w2 projection, reduce over N -> attention logits ----
        // C/D layout: N = lane&15; VGPR r -> row M = r + 8*(lane>=16).
        const int rowoff = (lane >> 4) << 3;
#pragma unroll
        for (int r = 0; r < 8; ++r) {
            float s = fast_tanhf_(acc0[r] + b1A) * w2A + fast_tanhf_(acc1[r] + b1B) * w2B;
            s += __shfl_xor(s, 1, 32);
            s += __shfl_xor(s, 2, 32);
            s += __shfl_xor(s, 4, 32);
            s += __shfl_xor(s, 8, 32);
            if ((lane & 15) == 0) {
                const int pair = wave * 16 + r + rowoff;
                lg[pair >> 5][pair & 31] = s + b2;
            }
        }
        __syncthreads();

        // ---- phase 4: softmax over memories, o = sum t*g, GRU cell ----
        if (wave < NB) {
            const int bl = wave;
            float l = lg[bl][lane];
            float mx = l;
#pragma unroll
            for (int d = 16; d >= 1; d >>= 1) mx = fmaxf(mx, __shfl_xor(mx, d, 32));
            float e  = __expf(l - mx);
            float se = e;
#pragma unroll
            for (int d = 16; d >= 1; d >>= 1) se += __shfl_xor(se, d, 32);
            const float g = e / se;

            float o = 0.0f;
#pragma unroll
            for (int m = 0; m < NMEM; ++m) o += tS[bl][m][lane] * __shfl(g, m, 32);

            const float* wih = gru_w_ih + hop * 3 * DIM * DIM;
            const float* whh = gru_w_hh + hop * 3 * DIM * DIM;
            const float  Mv  = MS[bl][lane];
            float gir = gru_b_ih[hop * 3 * DIM + lane];
            float giz = gru_b_ih[hop * 3 * DIM + DIM + lane];
            float gin = gru_b_ih[hop * 3 * DIM + 2 * DIM + lane];
            float ghr = gru_b_hh[hop * 3 * DIM + lane];
            float ghz = gru_b_hh[hop * 3 * DIM + DIM + lane];
            float ghn = gru_b_hh[hop * 3 * DIM + 2 * DIM + lane];
            for (int j = 0; j < DIM; ++j) {
                const float xj = __shfl(o, j, 32);
                const float hj = __shfl(Mv, j, 32);
                gir += wih[lane * DIM + j] * xj;
                giz += wih[(DIM + lane) * DIM + j] * xj;
                gin += wih[(2 * DIM + lane) * DIM + j] * xj;
                ghr += whh[lane * DIM + j] * hj;
                ghz += whh[(DIM + lane) * DIM + j] * hj;
                ghn += whh[(2 * DIM + lane) * DIM + j] * hj;
            }
            const float rr = sigmoidf_(gir + ghr);
            const float zz = sigmoidf_(giz + ghz);
            const float nn = fast_tanhf_(gin + rr * ghn);
            MS[bl][lane] = (1.0f - zz) * nn + zz * Mv;
        }
        __syncthreads();
    }

    // ---- final readout: out[b] = sum_i v[i] * sigmoid((M @ ans_w^T + ans_b)[i]) ----
    if (wave < NB) {
        const int bl = wave;
        const int bg = blockIdx.x * NB + bl;
        const float Mv = MS[bl][lane];
        float a = ans_b[lane];
        for (int j = 0; j < DIM; ++j) a += ans_w[lane * DIM + j] * __shfl(Mv, j, 32);
        float p = vS[bl][lane] * sigmoidf_(a);
#pragma unroll
        for (int d = 16; d >= 1; d >>= 1) p += __shfl_xor(p, d, 32);
        if (lane == 0) out[bg] = p;
    }
}

extern "C" void kernel_launch(void* const* d_in, const int* in_sizes, int n_in,
                              void* d_out, int out_size, void* d_ws, size_t ws_size,
                              hipStream_t stream) {
    const int*   h_i          = (const int*)d_in[0];
    const int*   R_i          = (const int*)d_in[1];
    const int*   t_i          = (const int*)d_in[2];
    const int*   v_i          = (const int*)d_in[3];
    const float* entity_emb   = (const float*)d_in[4];
    const float* relation_emb = (const float*)d_in[5];
    const float* agg_w1       = (const float*)d_in[6];
    const float* agg_b1       = (const float*)d_in[7];
    const float* agg_w2       = (const float*)d_in[8];
    const float* agg_b2       = (const float*)d_in[9];
    const float* gru_w_ih     = (const float*)d_in[10];
    const float* gru_w_hh     = (const float*)d_in[11];
    const float* gru_b_ih     = (const float*)d_in[12];
    const float* gru_b_hh     = (const float*)d_in[13];
    const float* ans_w        = (const float*)d_in[14];
    const float* ans_b        = (const float*)d_in[15];
    float*       out          = (float*)d_out;

    const int BATCH = 2048;
    dim3 grid(BATCH / NB);
    dim3 block(256);
    ripplenet_kernel<<<grid, block, 0, stream>>>(
        h_i, R_i, t_i, v_i, entity_emb, relation_emb,
        agg_w1, agg_b1, agg_w2, agg_b2,
        gru_w_ih, gru_w_hh, gru_b_ih, gru_b_hh,
        ans_w, ans_b, out);
}
